// PointwiseAggregatedAttention_90125593740076
// MI455X (gfx1250) — matrix operand order repaired and checked by reading, not compiled, not run on hardware
//
#include <hip/hip_runtime.h>

typedef __attribute__((ext_vector_type(16))) __bf16 v16bf;
typedef __attribute__((ext_vector_type(8)))  __bf16 v8bf;
typedef __attribute__((ext_vector_type(8)))  float  v8f;

#define WMMA_BF16(a,b,c) \
  __builtin_amdgcn_wmma_f32_16x16x32_bf16(false,(a),false,(b),(short)0,(c),false,false)

constexpr int S  = 2048;
constexpr int D  = 1024;
constexpr int H  = 16;
constexpr int HD = 64;
constexpr int B  = 2;
constexpr int BQ = 64;   // query rows per block (4 waves x 16)
constexpr int BK = 32;   // keys per iteration (K-dim of 2nd WMMA)
constexpr int QP = 72;   // LDS pitches (elements); multiples of 8 -> 16B aligned rows
constexpr int KP = 72;
constexpr int VP = 40;
constexpr int WP = 40;
constexpr float SCALE = 0.125f;                       // 1/sqrt(64)
constexpr float BUCKET_SCL = 16.0f / 2.0794415416798357f; // 16/log(8)

// A-operand (16-bit, 16x32): lanes 0-15 hold K={0..7,16..23}, lanes 16-31 K={8..15,24..31}
__device__ __forceinline__ v16bf load_a32(const __bf16* w, int hlf) {
  v8bf lo = *(const v8bf*)(w + 8 * hlf);
  v8bf hi = *(const v8bf*)(w + 16 + 8 * hlf);
  return __builtin_shufflevector(lo, hi, 0,1,2,3,4,5,6,7,8,9,10,11,12,13,14,15);
}
// B-operand (16-bit, 32x16) read from row-major B^T tile: lanes 0-15 hold K={0..7,16..23},
// lanes 16-31 hold K={8..15,24..31} of column n = lane%16
__device__ __forceinline__ v16bf load_b32(const __bf16* w, int hlf) {
  v8bf lo = *(const v8bf*)(w + 8 * hlf);
  v8bf hi = *(const v8bf*)(w + 16 + 8 * hlf);
  return __builtin_shufflevector(lo, hi, 0,1,2,3,4,5,6,7,8,9,10,11,12,13,14,15);
}

__device__ __forceinline__ float softplus_f(float x) {
  return fmaxf(x, 0.0f) + __logf(1.0f + __expf(-fabsf(x)));
}

__global__ __launch_bounds__(128, 1)
void paa_fused_kernel(const float* __restrict__ Vg, const float* __restrict__ Kg,
                      const float* __restrict__ Qg, const int*   __restrict__ Mg,
                      const float* __restrict__ RBg, float* __restrict__ Og)
{
  __shared__ __bf16 Qs [BQ * QP];        // Q tile, row-major [q][dk]
  __shared__ __bf16 Ks [BK * KP];        // K tile, row-major [key][dk]  (== B^T for QK^T)
  __shared__ __bf16 VTs[HD * VP];        // V tile transposed [dv][key]  (== B^T for W·V)
  __shared__ __bf16 Ws [4 * 16 * WP];    // per-wave W scratch [qrow][key]
  __shared__ float  biasS[S];            // rel-position bias LUT for this head
  __shared__ int    kmS[BK];

  const int q0   = blockIdx.x * BQ;
  const int bh   = blockIdx.y;
  const int b    = bh >> 4;
  const int h    = bh & 15;
  const int tid  = threadIdx.x;
  const int wave = tid >> 5;
  const int lane = tid & 31;
  const int hlf  = lane >> 4;
  const int r    = lane & 15;

  // ---- per-head relative-position bias LUT: biasS[d] = rel_bias[bucket(d), h] ----
  for (int d = tid; d < S; d += 128) {
    int bkt;
    if (d < 16) bkt = d;
    else {
      bkt = 16 + (int)(__logf((float)d * (1.0f / 16.0f)) * BUCKET_SCL);
      bkt = min(bkt, 31);
    }
    biasS[d] = RBg[bkt * H + h];
  }

  // ---- stage Q tile (64 x 64) as bf16 ----
  {
    const float* qbase = Qg + ((size_t)b * S + q0) * D + h * HD;
    for (int i = tid; i < BQ * 16; i += 128) {
      int row = i >> 4, g = i & 15;
      const float4 f = *(const float4*)(qbase + (size_t)row * D + g * 4);
      __bf16* p = &Qs[row * QP + g * 4];
      p[0] = (__bf16)f.x; p[1] = (__bf16)f.y; p[2] = (__bf16)f.z; p[3] = (__bf16)f.w;
    }
  }

  const int q0w = q0 + wave * 16;          // this wave's q rows: q0w .. q0w+15
  unsigned qm = 0;                         // query-mask bits for this lane's 8 rows
  #pragma unroll
  for (int vp = 0; vp < 8; ++vp)
    if (Mg[b * S + q0w + vp + 8 * hlf]) qm |= 1u << vp;

  v8f acc0 = {}, acc1 = {}, acc2 = {}, acc3 = {};   // out[16 x 64] fp32
  float den[8] = {0.f,0.f,0.f,0.f,0.f,0.f,0.f,0.f};

  const int nkb = q0 / BK + 2;             // causal: keys 0 .. q0+63
  for (int kb = 0; kb < nkb; ++kb) {
    const int kblk = kb * BK;
    __syncthreads();                       // prior compute done; Qs/biasS visible on kb==0

    // ---- stage K tile row-major and V tile transposed (32 keys x 64 ch) ----
    const float* kbase = Kg + ((size_t)b * S + kblk) * D + h * HD;
    const float* vbase = Vg + ((size_t)b * S + kblk) * D + h * HD;
    for (int i = tid; i < BK * 16; i += 128) {
      int row = i >> 4, g = i & 15;
      float4 f = *(const float4*)(kbase + (size_t)row * D + g * 4);
      __bf16* p = &Ks[row * KP + g * 4];
      p[0] = (__bf16)f.x; p[1] = (__bf16)f.y; p[2] = (__bf16)f.z; p[3] = (__bf16)f.w;
      f = *(const float4*)(vbase + (size_t)row * D + g * 4);
      VTs[(g * 4 + 0) * VP + row] = (__bf16)f.x;
      VTs[(g * 4 + 1) * VP + row] = (__bf16)f.y;
      VTs[(g * 4 + 2) * VP + row] = (__bf16)f.z;
      VTs[(g * 4 + 3) * VP + row] = (__bf16)f.w;
    }
    if (tid < BK) kmS[tid] = Mg[b * S + kblk + tid];
    __syncthreads();

    if (kblk <= q0w + 15) {                // wave-uniform causal skip (EXEC stays full)
      // ---- scores: two 16x16 tiles, K-dim 64 = 2 x WMMA(K=32) each ----
      v16bf aq0 = load_a32(&Qs[(wave * 16 + r) * QP + 0 ], hlf);
      v16bf aq1 = load_a32(&Qs[(wave * 16 + r) * QP + 32], hlf);
      v8f s0 = {}, s1 = {};
      {
        v16bf bk = load_b32(&Ks[r * KP + 0], hlf);
        s0 = WMMA_BF16(aq0, bk, s0);
        bk = load_b32(&Ks[r * KP + 32], hlf);
        s0 = WMMA_BF16(aq1, bk, s0);
        bk = load_b32(&Ks[(16 + r) * KP + 0], hlf);
        s1 = WMMA_BF16(aq0, bk, s1);
        bk = load_b32(&Ks[(16 + r) * KP + 32], hlf);
        s1 = WMMA_BF16(aq1, bk, s1);
      }

      // ---- mask + bias + softplus; spill W (bf16) to wave-private LDS in A layout ----
      __bf16* wrow = &Ws[wave * 16 * WP];
      const int km0 = kmS[r], km1 = kmS[16 + r];
      #pragma unroll
      for (int vp = 0; vp < 8; ++vp) {
        const int qabs = q0w + vp + 8 * hlf;
        {
          const int kabs = kblk + r;
          float w = 0.0f;
          if (kabs <= qabs && ((qm >> vp) & 1u) && km0)
            w = softplus_f(s0[vp] * SCALE + biasS[qabs - kabs]);
          den[vp] += w;
          wrow[(vp + 8 * hlf) * WP + r] = (__bf16)w;
        }
        {
          const int kabs = kblk + 16 + r;
          float w = 0.0f;
          if (kabs <= qabs && ((qm >> vp) & 1u) && km1)
            w = softplus_f(s1[vp] * SCALE + biasS[qabs - kabs]);
          den[vp] += w;
          wrow[(vp + 8 * hlf) * WP + 16 + r] = (__bf16)w;
        }
      }
      asm volatile("s_wait_dscnt 0" ::: "memory");   // wave-private LDS round trip

      // ---- out += W (16x32) x V (32x64), 4 N-tiles ----
      v16bf aw = load_a32(&wrow[r * WP], hlf);
      v16bf bv = load_b32(&VTs[r * VP], hlf);
      acc0 = WMMA_BF16(aw, bv, acc0);
      bv = load_b32(&VTs[(16 + r) * VP], hlf);
      acc1 = WMMA_BF16(aw, bv, acc1);
      bv = load_b32(&VTs[(32 + r) * VP], hlf);
      acc2 = WMMA_BF16(aw, bv, acc2);
      bv = load_b32(&VTs[(48 + r) * VP], hlf);
      acc3 = WMMA_BF16(aw, bv, acc3);
    }
  }

  // ---- denominator: reduce across the 16 lanes sharing this hlf (same q rows) ----
  float inv[8];
  #pragma unroll
  for (int vp = 0; vp < 8; ++vp) {
    float s = den[vp];
    s += __shfl_xor(s, 1, 32);
    s += __shfl_xor(s, 2, 32);
    s += __shfl_xor(s, 4, 32);
    s += __shfl_xor(s, 8, 32);
    inv[vp] = 1.0f / (s + 1e-8f);
  }

  // ---- write out[b, q, h*64 + dv] (fp32) ----
  float* obase = Og + (size_t)b * S * D + h * HD;
  #pragma unroll
  for (int vp = 0; vp < 8; ++vp) {
    const int qabs = q0w + vp + 8 * hlf;
    float* orow = obase + (size_t)qabs * D;
    orow[ 0 + r] = acc0[vp] * inv[vp];
    orow[16 + r] = acc1[vp] * inv[vp];
    orow[32 + r] = acc2[vp] * inv[vp];
    orow[48 + r] = acc3[vp] * inv[vp];
  }
}

extern "C" void kernel_launch(void* const* d_in, const int* in_sizes, int n_in,
                              void* d_out, int out_size, void* d_ws, size_t ws_size,
                              hipStream_t stream) {
  (void)in_sizes; (void)n_in; (void)out_size; (void)d_ws; (void)ws_size;
  const float* v  = (const float*)d_in[0];
  const float* k  = (const float*)d_in[1];
  const float* q  = (const float*)d_in[2];
  const int*   m  = (const int*)  d_in[3];
  const float* rb = (const float*)d_in[4];
  float*       o  = (float*)d_out;
  dim3 grid(S / BQ, B * H);
  paa_fused_kernel<<<grid, 128, 0, stream>>>(v, k, q, m, rb, o);
}
